// TemporalWormhole_56762287784383
// MI455X (gfx1250) — matrix operand order repaired and checked by reading, not compiled, not run on hardware
//
#include <hip/hip_runtime.h>

// ---------------------------------------------------------------------------
// Types for CDNA5 WMMA (wave32, v_wmma_f32_16x16x32_bf16)
// ---------------------------------------------------------------------------
typedef __bf16 bf16_t;
typedef __attribute__((ext_vector_type(16))) __bf16 v16bf;
typedef __attribute__((ext_vector_type(8)))  __bf16 v8bf;
typedef __attribute__((ext_vector_type(8)))  float  v8f;
typedef __attribute__((ext_vector_type(4)))  int    v4i_t;

union V16U { v16bf v; v8bf h[2]; };

// gfx1250 async global->LDS path (ASYNCcnt-tracked, no VGPR staging).
// The builtin exists on this toolchain; its signature (leaked by the round-2
// diagnostic) is (v4i addrspace(1)*, v4i addrspace(3)*, imm offset, imm cpol).
#if defined(__HIP_DEVICE_COMPILE__) &&                                   \
    __has_builtin(__builtin_amdgcn_global_load_async_to_lds_b128) &&     \
    __has_builtin(__builtin_amdgcn_s_wait_asynccnt)
#define USE_ASYNC_LDS 1
#else
#define USE_ASYNC_LDS 0
#endif

typedef __attribute__((address_space(1))) v4i_t as1_v4i;
typedef __attribute__((address_space(3))) v4i_t as3_v4i;

__device__ inline void copy16_g2l(bf16_t* lds, const bf16_t* glb) {
#if USE_ASYNC_LDS
  __builtin_amdgcn_global_load_async_to_lds_b128(
      (as1_v4i*)(void*)const_cast<bf16_t*>(glb), (as3_v4i*)lds, 0, 0);
#else
  *(v8bf*)lds = *(const v8bf*)glb;
#endif
}

__device__ inline void async_copy_wait() {
#if USE_ASYNC_LDS
  __builtin_amdgcn_s_wait_asynccnt(0);
#endif
}

__device__ inline v8f wmma_bf16(v16bf a, v16bf b, v8f c) {
  // 8 args: (neg_a, A, neg_b, B, c_mod, C, reuse_a, reuse_b)
  return __builtin_amdgcn_wmma_f32_16x16x32_bf16(false, a, false, b, (short)0, c,
                                                 false, false);
}

// A-fragment (16x32 bf16) from row-major [16][stride] source.
// lane 0-15 : row = lane,    K = 0..15
// lane 16-31: row = lane-16, K = 16..31
__device__ inline v16bf ld_afrag(const bf16_t* base, int stride, int lane) {
  int r = lane & 15, hs = (lane >> 4) & 1;
  const bf16_t* p = base + (size_t)r * stride;
  V16U u;
  u.h[0] = *(const v8bf*)(p + hs * 16);
  u.h[1] = *(const v8bf*)(p + hs * 16 + 8);
  return u.v;
}

// B-fragment (32x16 bf16) from Bt rows (row = N column, K contiguous).
// lane 0-15 : N = lane,    K = 0..15 ; lane 16-31: N = lane-16, K = 16..31
__device__ inline v16bf ld_bfrag(const bf16_t* base, int stride, int lane) {
  int r = lane & 15, hs = (lane >> 4) & 1;
  const bf16_t* p = base + (size_t)r * stride + hs * 16;
  V16U u;
  u.h[0] = *(const v8bf*)(p);
  u.h[1] = *(const v8bf*)(p + 8);
  return u.v;
}

// ---------------------------------------------------------------------------
// Tiled bf16 GEMM:  C[M,N](f32/bf16) = A[M,K] @ Bt[N,K]^T + bias (+ C if acc)
// Block tile 128x128, 8 waves each computing 64x32 (4x2 WMMA accum tiles),
// double-buffered LDS fed by async global->LDS, K step 32.
// LDS rows padded to 48 elements (96B) to cut ds_load_b128 bank conflicts.
// M%128==0, N%128==0, K%32==0 assumed.
// ---------------------------------------------------------------------------
#define GBM 128
#define GBN 128
#define GBK 32
#define ASTR 48  // padded LDS row stride (elements)

__global__ __launch_bounds__(256) void gemm_bf16_kernel(
    const bf16_t* __restrict__ A, int lda,
    const bf16_t* __restrict__ Bt, int ldb,
    const float* __restrict__ bias,
    float* __restrict__ Cf, bf16_t* __restrict__ Cb,
    int K, int ldc, int accLoad) {
  __shared__ __align__(16) bf16_t As[2][GBM * ASTR];
  __shared__ __align__(16) bf16_t Bs[2][GBN * ASTR];

  const int tid  = threadIdx.x;
  const int lane = tid & 31;
  const int wave = tid >> 5;
  const int bm = blockIdx.x * GBM;
  const int bn = blockIdx.y * GBN;
  const int wrow = (wave & 1) * 64;   // 2 wave rows of 64
  const int wcol = (wave >> 1) * 32;  // 4 wave cols of 32

  // Per-thread copy assignment: row r0 and r0+64, one 16B chunk each, A and B.
  const int r0 = tid >> 2;           // 0..63
  const int kc = (tid & 3) * 8;      // 0,8,16,24
  const bf16_t* gA0 = A + (size_t)(bm + r0) * lda + kc;
  const bf16_t* gA1 = gA0 + (size_t)64 * lda;
  const bf16_t* gB0 = Bt + (size_t)(bn + r0) * ldb + kc;
  const bf16_t* gB1 = gB0 + (size_t)64 * ldb;

  v8f acc[4][2] = {};

  auto loadTile = [&](int buf, int k0) {
    bf16_t* sA = &As[buf][r0 * ASTR + kc];
    bf16_t* sB = &Bs[buf][r0 * ASTR + kc];
    copy16_g2l(sA, gA0 + k0);
    copy16_g2l(sA + 64 * ASTR, gA1 + k0);
    copy16_g2l(sB, gB0 + k0);
    copy16_g2l(sB + 64 * ASTR, gB1 + k0);
    if (k0 + 2 * GBK < K) {  // warm L2/WGP$ two tiles ahead
      __builtin_prefetch(gA0 + k0 + 2 * GBK, 0, 0);
      __builtin_prefetch(gB0 + k0 + 2 * GBK, 0, 0);
    }
  };

  loadTile(0, 0);
  async_copy_wait();
  __syncthreads();
  const int KT = K / GBK;
  for (int kt = 0; kt < KT; ++kt) {
    if (kt + 1 < KT) loadTile((kt + 1) & 1, (kt + 1) * GBK);
    const bf16_t* Ab = &As[kt & 1][0];
    const bf16_t* Bb = &Bs[kt & 1][0];
    v16bf aF[4], bF[2];
#pragma unroll
    for (int r = 0; r < 4; ++r) aF[r] = ld_afrag(Ab + (wrow + r * 16) * ASTR, ASTR, lane);
#pragma unroll
    for (int c = 0; c < 2; ++c) bF[c] = ld_bfrag(Bb + (wcol + c * 16) * ASTR, ASTR, lane);
#pragma unroll
    for (int r = 0; r < 4; ++r)
#pragma unroll
      for (int c = 0; c < 2; ++c) acc[r][c] = wmma_bf16(aF[r], bF[c], acc[r][c]);
    async_copy_wait();
    __syncthreads();
  }

  // Epilogue. C layout: lane 0-15 -> N=lane, rows j=0..7; lane 16-31 -> N=lane-16, rows 8+j.
  const int hs = (lane >> 4) & 1, ln = lane & 15;
#pragma unroll
  for (int r = 0; r < 4; ++r) {
#pragma unroll
    for (int c = 0; c < 2; ++c) {
      int col = bn + wcol + c * 16 + ln;
      float bv = bias ? bias[col] : 0.0f;
      int rowbase = bm + wrow + r * 16 + hs * 8;
#pragma unroll
      for (int j = 0; j < 8; ++j) {
        size_t idx = (size_t)(rowbase + j) * ldc + col;
        float v = acc[r][c][j] + bv;
        if (accLoad) v += Cf[idx];
        if (Cf) Cf[idx] = v;
        if (Cb) Cb[idx] = (bf16_t)v;
      }
    }
  }
}

// ---------------------------------------------------------------------------
// Flash-style windowed causal attention, one wave = 16 queries, full D=256.
// q,k: [band*8+b][T][D] bf16 row-major. vT: [band*8+b][D][T] bf16.
// ctx out: bf16 [band*8+b][T][D].
// ---------------------------------------------------------------------------
__constant__ int c_win[7] = {128, 64, 32, 16, 16, 8, 4};

__global__ __launch_bounds__(256) void attn_kernel(
    const bf16_t* __restrict__ q, const bf16_t* __restrict__ k,
    const bf16_t* __restrict__ vT, bf16_t* __restrict__ ctx) {
  __shared__ __align__(16) bf16_t Pl[8][16 * ASTR];

  const int band = blockIdx.z, b = blockIdx.y, qblk = blockIdx.x;
  const int tid = threadIdx.x, lane = tid & 31, wave = tid >> 5;
  const int qt0 = qblk * 128 + wave * 16;
  const int w = c_win[band];
  const size_t base = ((size_t)(band * 8 + b)) * 1024 * 256;
  const bf16_t* qp = q + base;
  const bf16_t* kp = k + base;
  const bf16_t* vp = vT + base;  // [256][1024] per (band,b)
  bf16_t* cp = ctx + base;
  const int hs = (lane >> 4) & 1, ln = lane & 15;

  float m[8], l[8];
  v8f o[16] = {};
#pragma unroll
  for (int j = 0; j < 8; ++j) { m[j] = -3.0e38f; l[j] = 0.0f; }

  int kbeg = qt0 + 1 - w;
  if (kbeg < 0) kbeg = 0;
  kbeg &= ~31;
  const int kend = qt0 + 15;

  for (int kb = kbeg; kb <= kend; kb += 32) {
    v8f S0 = {}, S1 = {};
#pragma unroll
    for (int dk = 0; dk < 8; ++dk) {
      v16bf qa = ld_afrag(qp + (size_t)qt0 * 256 + dk * 32, 256, lane);
      v16bf k0 = ld_bfrag(kp + (size_t)kb * 256 + dk * 32, 256, lane);
      v16bf k1 = ld_bfrag(kp + (size_t)(kb + 16) * 256 + dk * 32, 256, lane);
      S0 = wmma_bf16(qa, k0, S0);
      S1 = wmma_bf16(qa, k1, S1);
    }

    float p0[8], p1[8];
#pragma unroll
    for (int j = 0; j < 8; ++j) {
      int qrow = qt0 + j + hs * 8;
      int key0 = kb + ln, key1 = kb + 16 + ln;
      float s0 = (key0 <= qrow && key0 > qrow - w) ? S0[j] * 0.0625f : -3.0e38f;
      float s1 = (key1 <= qrow && key1 > qrow - w) ? S1[j] * 0.0625f : -3.0e38f;
      float bm_ = fmaxf(s0, s1);
#pragma unroll
      for (int mk = 1; mk < 16; mk <<= 1) bm_ = fmaxf(bm_, __shfl_xor(bm_, mk, 32));
      float mn = fmaxf(m[j], bm_);
      float al = __expf(m[j] - mn);
      p0[j] = __expf(s0 - mn);
      p1[j] = __expf(s1 - mn);
      float rs = p0[j] + p1[j];
#pragma unroll
      for (int mk = 1; mk < 16; mk <<= 1) rs += __shfl_xor(rs, mk, 32);
      l[j] = l[j] * al + rs;
      m[j] = mn;
#pragma unroll
      for (int t = 0; t < 16; ++t) o[t][j] *= al;
    }

    // C-layout P tile -> LDS (wave-private, padded rows) -> A-layout fragment
    bf16_t* pl = &Pl[wave][0];
#pragma unroll
    for (int j = 0; j < 8; ++j) {
      pl[(j + hs * 8) * ASTR + ln]      = (bf16_t)p0[j];
      pl[(j + hs * 8) * ASTR + 16 + ln] = (bf16_t)p1[j];
    }
    v16bf pf = ld_afrag(pl, ASTR, lane);
#pragma unroll
    for (int n = 0; n < 16; ++n) {
      v16bf vf = ld_bfrag(vp + (size_t)(n * 16) * 1024 + kb, 1024, lane);
      o[n] = wmma_bf16(pf, vf, o[n]);
    }
  }

  float invl[8];
#pragma unroll
  for (int j = 0; j < 8; ++j) invl[j] = 1.0f / l[j];
#pragma unroll
  for (int n = 0; n < 16; ++n)
#pragma unroll
    for (int j = 0; j < 8; ++j)
      cp[(size_t)(qt0 + j + hs * 8) * 256 + n * 16 + ln] = (bf16_t)(o[n][j] * invl[j]);
}

// ---------------------------------------------------------------------------
// Elementwise helpers
// ---------------------------------------------------------------------------
__global__ void cvt_f32_bf16_kernel(const float* __restrict__ in,
                                    bf16_t* __restrict__ out, long n) {
  for (long i = (long)blockIdx.x * blockDim.x + threadIdx.x; i < n;
       i += (long)gridDim.x * blockDim.x)
    out[i] = (bf16_t)in[i];
}

// v [nb][1024][256] -> vT [nb][256][1024]
__global__ void transpose_v_kernel(const bf16_t* __restrict__ v,
                                   bf16_t* __restrict__ vt, long n) {
  long i = (long)blockIdx.x * blockDim.x + threadIdx.x;
  if (i >= n) return;
  long nb = i >> 18;
  int r = (int)(i & 262143);
  int d = r >> 10, t = r & 1023;
  vt[i] = v[(nb << 18) + (size_t)t * 256 + d];
}

__global__ void avg_other_kernel(const bf16_t* __restrict__ ctx,
                                 bf16_t* __restrict__ out, int n) {
  int i = blockIdx.x * blockDim.x + threadIdx.x;
  if (i >= n) return;
  float s = 0.0f;
#pragma unroll
  for (int band = 0; band < 7; ++band)
    if (band != 3) s += (float)ctx[(size_t)band * 2097152 + i];
  out[i] = (bf16_t)(s * (1.0f / 6.0f));
}

__global__ void gate_apply_kernel(float* __restrict__ out,
                                  const float* __restrict__ acc,
                                  const float* __restrict__ wf, int n) {
  int i = blockIdx.x * blockDim.x + threadIdx.x;
  if (i >= n) return;
  float g = 1.0f / (1.0f + __expf(-acc[i]));
  out[i] += g * wf[i];
}

// ---------------------------------------------------------------------------
// Host orchestration
// ---------------------------------------------------------------------------
extern "C" void kernel_launch(void* const* d_in, const int* in_sizes, int n_in,
                              void* d_out, int out_size, void* d_ws, size_t ws_size,
                              hipStream_t stream) {
  (void)in_sizes; (void)n_in; (void)out_size; (void)ws_size;
  const size_t BTD = 8ull * 1024 * 256;  // 2,097,152 per band
  const size_t NBD = 7 * BTD;            // 14,680,064
  const size_t DD  = 256 * 256;          // 65,536

  const float* band_in[7];
  for (int i = 0; i < 7; ++i) band_in[i] = (const float*)d_in[i];
  const float* qw      = (const float*)d_in[7];
  const float* qb      = (const float*)d_in[8];
  const float* kw      = (const float*)d_in[9];
  const float* kb_     = (const float*)d_in[10];
  const float* vw      = (const float*)d_in[11];
  const float* vb      = (const float*)d_in[12];
  const float* crossw  = (const float*)d_in[13];
  const float* crossb  = (const float*)d_in[14];
  const float* crossrw = (const float*)d_in[15];
  const float* crossrb = (const float*)d_in[16];
  const float* gatew   = (const float*)d_in[17];
  const float* gateb   = (const float*)d_in[18];
  const float* bridgew = (const float*)d_in[19];
  const float* bridgeb = (const float*)d_in[20];
  float* out = (float*)d_out;

  char* ws = (char*)d_ws;
  size_t off = 0;
  auto take = [&](size_t bytes) -> void* {
    void* p = ws + off;
    off = (off + bytes + 255) & ~(size_t)255;
    return p;
  };
  // bf16 activations / weights (64KB pads cover masked OOB key reads)
  bf16_t* bandsbf = (bf16_t*)take(NBD * 2);
  bf16_t* qbf     = (bf16_t*)take(NBD * 2 + 65536);
  bf16_t* kbf     = (bf16_t*)take(NBD * 2 + 65536);
  bf16_t* vbf     = (bf16_t*)take(NBD * 2 + 65536);
  bf16_t* vtbf    = (bf16_t*)take(NBD * 2 + 65536);
  bf16_t* ctxbf   = (bf16_t*)take(NBD * 2 + 65536);
  bf16_t* qwbf    = (bf16_t*)take(7 * DD * 2);
  bf16_t* kwbf    = (bf16_t*)take(7 * DD * 2);
  bf16_t* vwbf    = (bf16_t*)take(7 * DD * 2);
  bf16_t* crwbf   = (bf16_t*)take(3 * DD * 2);
  bf16_t* crrwbf  = (bf16_t*)take(3 * DD * 2);
  bf16_t* gwbf    = (bf16_t*)take(7 * 2 * DD * 2);  // [7][256][512]
  bf16_t* brwbf   = (bf16_t*)take(DD * 2);
  float*  w2f     = (float*) take(6 * BTD * 4);     // [p*2 + {0:high,1:low}]
  bf16_t* w2b     = (bf16_t*)take(6 * BTD * 2);
  float*  accf    = (float*) take(7 * BTD * 4);     // per-band gate pre-activation
  bf16_t* otherbf = (bf16_t*)take(BTD * 2);
  float*  brgf    = (float*) take(BTD * 4);
  bf16_t* brgb    = (bf16_t*)take(BTD * 2);

  auto cvt = [&](const float* in, bf16_t* o, size_t n) {
    int blocks = (int)((n + 255) / 256);
    if (blocks > 16384) blocks = 16384;
    cvt_f32_bf16_kernel<<<blocks, 256, 0, stream>>>(in, o, (long)n);
  };
  auto gemm = [&](const bf16_t* A, int lda, const bf16_t* Bt, int ldb,
                  const float* bias, float* Cf, bf16_t* Cb, int accLoad) {
    dim3 grid(8192 / GBM, 256 / GBN);  // M=8192, N=256, K=256 for all GEMMs here
    gemm_bf16_kernel<<<grid, 256, 0, stream>>>(A, lda, Bt, ldb, bias, Cf, Cb,
                                               256, 256, accLoad);
  };

  // 0) outputs start as the input bands
  for (int i = 0; i < 7; ++i)
    (void)hipMemcpyAsync(out + i * BTD, band_in[i], BTD * 4,
                         hipMemcpyDeviceToDevice, stream);

  // 1) demote everything to bf16
  for (int i = 0; i < 7; ++i) cvt(band_in[i], bandsbf + i * BTD, BTD);
  cvt(qw, qwbf, 7 * DD);   cvt(kw, kwbf, 7 * DD);   cvt(vw, vwbf, 7 * DD);
  cvt(crossw, crwbf, 3 * DD); cvt(crossrw, crrwbf, 3 * DD);
  cvt(gatew, gwbf, 7 * 2 * DD); cvt(bridgew, brwbf, DD);

  // 2) QKV projections (weights' [out][in] layout == WMMA Bt layout)
  for (int n = 0; n < 7; ++n) {
    gemm(bandsbf + n * BTD, 256, qwbf + n * DD, 256, qb + n * 256, nullptr, qbf + n * BTD, 0);
    gemm(bandsbf + n * BTD, 256, kwbf + n * DD, 256, kb_ + n * 256, nullptr, kbf + n * BTD, 0);
    gemm(bandsbf + n * BTD, 256, vwbf + n * DD, 256, vb + n * 256, nullptr, vbf + n * BTD, 0);
  }

  // 3) v -> vT for the attn@v WMMA B operand
  {
    long n = (long)NBD;
    transpose_v_kernel<<<(int)((n + 255) / 256), 256, 0, stream>>>(vbf, vtbf, n);
  }

  // 4) windowed causal flash attention (all 7 bands in one launch)
  {
    dim3 grid(8 /*q-blocks of 128*/, 8 /*batch*/, 7 /*band*/);
    attn_kernel<<<grid, 256, 0, stream>>>(qbf, kbf, vtbf, ctxbf);
  }

  // 5) cross-band projections + gates + output updates
  const int lo[3] = {0, 1, 2}, hi[3] = {6, 5, 4};
  for (int p = 0; p < 3; ++p) {
    float*  w2hf = w2f + (size_t)(2 * p) * BTD;
    bf16_t* w2hb = w2b + (size_t)(2 * p) * BTD;
    float*  w2lf = w2f + (size_t)(2 * p + 1) * BTD;
    bf16_t* w2lb = w2b + (size_t)(2 * p + 1) * BTD;
    gemm(ctxbf + lo[p] * BTD, 256, crwbf + p * DD, 256, crossb + p * 256, w2hf, w2hb, 0);
    gemm(ctxbf + hi[p] * BTD, 256, crrwbf + p * DD, 256, crossrb + p * 256, w2lf, w2lb, 0);
    // gate(high) = sigmoid([band_high | w2h] @ gatew_high^T + b): split into two GEMMs
    gemm(bandsbf + hi[p] * BTD, 256, gwbf + (size_t)hi[p] * 2 * DD, 512,
         gateb + hi[p] * 256, accf + hi[p] * BTD, nullptr, 0);
    gemm(w2hb, 256, gwbf + (size_t)hi[p] * 2 * DD + 256, 512,
         nullptr, accf + hi[p] * BTD, nullptr, 1);
    gemm(bandsbf + lo[p] * BTD, 256, gwbf + (size_t)lo[p] * 2 * DD, 512,
         gateb + lo[p] * 256, accf + lo[p] * BTD, nullptr, 0);
    gemm(w2lb, 256, gwbf + (size_t)lo[p] * 2 * DD + 256, 512,
         nullptr, accf + lo[p] * BTD, nullptr, 1);
    gate_apply_kernel<<<(int)(BTD / 256), 256, 0, stream>>>(
        out + hi[p] * BTD, accf + hi[p] * BTD, w2hf, (int)BTD);
    gate_apply_kernel<<<(int)(BTD / 256), 256, 0, stream>>>(
        out + lo[p] * BTD, accf + lo[p] * BTD, w2lf, (int)BTD);
  }

  // 6) bridge for band 3
  avg_other_kernel<<<(int)(BTD / 256), 256, 0, stream>>>(ctxbf, otherbf, (int)BTD);
  gemm(otherbf, 256, brwbf, 256, bridgeb, brgf, brgb, 0);
  gemm(bandsbf + 3 * BTD, 256, gwbf + (size_t)3 * 2 * DD, 512,
       gateb + 3 * 256, accf + 3 * BTD, nullptr, 0);
  gemm(brgb, 256, gwbf + (size_t)3 * 2 * DD + 256, 512,
       nullptr, accf + 3 * BTD, nullptr, 1);
  gate_apply_kernel<<<(int)(BTD / 256), 256, 0, stream>>>(
      out + 3 * BTD, accf + 3 * BTD, brgf, (int)BTD);
}
